// UpdateNet_26508538151194
// MI455X (gfx1250) — matrix operand order repaired and checked by reading, not compiled, not run on hardware
//
#include <hip/hip_runtime.h>

typedef __attribute__((ext_vector_type(16))) __bf16 v16bf;
typedef __attribute__((ext_vector_type(8)))  __bf16 v8bf;
typedef __attribute__((ext_vector_type(8)))  float  v8f;

constexpr int Bn = 16, C = 16, Hn = 256, Wn = 256, DN = 4, F1 = 52, HID = 128;
constexpr int WCH  = 16;  // pixels per row-segment (one 16-px WMMA tile)
constexpr int ROWS = 8;   // output rows per block (one per wave)

// ---------------- one-shot weight -> bf16 A-fragment prep (into d_ws) -------------
// w1 frags: [m(8)][kk(2)][lane(32)][h(16)]   (8192 elems)
// w2 frags: [kk(4)][lane(32)][h(16)]         (2048 elems, offset 8192)
// A-layout: K = 32*kk + 16*(lane>=16) + h ; M = 16*m + lane%16
__global__ __launch_bounds__(256) void prep_weights_kernel(
    const float* __restrict__ w1, const float* __restrict__ w2, __bf16* __restrict__ wf)
{
  int i = blockIdx.x * 256 + threadIdx.x;
  if (i < 8192) {
    int h = i & 15, l = (i >> 4) & 31, kk = (i >> 9) & 1, m = i >> 10;
    int K = kk * 32 + ((l >> 4) & 1) * 16 + h;
    int row = m * 16 + (l & 15);
    float v = (K < F1) ? w1[row * F1 + K] : 0.f;
    wf[i] = (__bf16)v;
  } else if (i < 10240) {
    int j = i - 8192;
    int h = j & 15, l = (j >> 4) & 31, kk = (j >> 9) & 3;
    int K = kk * 32 + ((l >> 4) & 1) * 16 + h;
    wf[i] = (__bf16)w2[(l & 15) * HID + K];
  }
}

// ---------------- fused NCA update ------------------------------------------------
struct alignas(32) SMem {
  float xt[16][ROWS + 2][20];          // x halo: ch, rows y0-1..y0+8, 18 cols used (pad 20)
  float dnat[ROWS][4][16];             // dna rows
  unsigned short hS[8][16][128];       // per-wave hidden panel [pixel][row128]
  unsigned short w1s[8192];            // W1 bf16 A-fragments staged in LDS (16 KB)
};                                     // total 64000 B

__global__ __launch_bounds__(256) void nca_update_kernel(
    const float* __restrict__ x, const float* __restrict__ dna,
    const __bf16* __restrict__ wf, float* __restrict__ out)
{
  __shared__ SMem sm;

  const int tid = threadIdx.x;
  const int bid = blockIdx.x;
  const int xc = bid & 15;                    // 16-px chunk of the row
  const int yp = (bid >> 4) & (Hn / ROWS - 1);
  const int b  = bid >> 9;
  const int x0 = xc * WCH;
  const int y0 = yp * ROWS;

  // ---- stage W1 fragments into LDS (16 KB, coalesced b128 copies) ----
  {
    const uint4* src = reinterpret_cast<const uint4*>(wf);
    uint4* dst = reinterpret_cast<uint4*>(sm.w1s);
#pragma unroll
    for (int i = 0; i < 4; ++i) dst[tid + i * 256] = src[tid + i * 256];
  }
  // ---- stage x halo tile (16ch x 10rows x 18cols, zero-padded) ----
  for (int i = tid; i < 16 * (ROWS + 2) * 18; i += 256) {
    int c = i / ((ROWS + 2) * 18), rem = i % ((ROWS + 2) * 18);
    int rr = rem / 18, jj = rem % 18;
    int yy = y0 + rr - 1, xx = x0 - 1 + jj;
    float v = 0.f;
    if (yy >= 0 && yy < Hn && xx >= 0 && xx < Wn)
      v = x[((b * C + c) * Hn + yy) * Wn + xx];
    sm.xt[c][rr][jj] = v;
  }
  // ---- stage dna rows ----
  for (int i = tid; i < ROWS * DN * WCH; i += 256) {
    int r = i >> 6, c = (i >> 4) & 3, p = i & 15;
    sm.dnat[r][c][p] = dna[((b * DN + c) * Hn + y0 + r) * Wn + x0 + p];
  }
  __syncthreads();

  const int wave = tid >> 5, lane = tid & 31, lp = lane & 15, hi = lane >> 4;
  const int r = wave;                 // this wave's output row
  const int j = lp + 1;               // center column in halo tile

  // ---- Phase A: build layer-1 B fragments directly in registers ----
  // B layout: lane holds pixel lp; element h = channel (16*hi + h) [+32 for b1]
  //   lanes 0-15 : b0 = x[0..15],      b1 = grad_y[0..15]
  //   lanes 16-31: b0 = grad_x[0..15], b1 = dna[0..3] + zero pad
  v16bf b0, b1;
  if (hi == 0) {
#pragma unroll
    for (int c = 0; c < 16; ++c) {
      b0[c] = (__bf16)sm.xt[c][r + 1][j];
      float gy = (sm.xt[c][r + 2][j - 1] + 2.f * sm.xt[c][r + 2][j] + sm.xt[c][r + 2][j + 1])
               - (sm.xt[c][r + 0][j - 1] + 2.f * sm.xt[c][r + 0][j] + sm.xt[c][r + 0][j + 1]);
      b1[c] = (__bf16)gy;
    }
  } else {
#pragma unroll
    for (int c = 0; c < 16; ++c) {
      float gx = (sm.xt[c][r + 0][j + 1] - sm.xt[c][r + 0][j - 1])
         + 2.f * (sm.xt[c][r + 1][j + 1] - sm.xt[c][r + 1][j - 1])
               + (sm.xt[c][r + 2][j + 1] - sm.xt[c][r + 2][j - 1]);
      b0[c] = (__bf16)gx;
      b1[c] = (c < DN) ? (__bf16)sm.dnat[r][c][lp] : (__bf16)0.0f;
    }
  }

  // ---- Phase B1: layer-1 WMMA stream, 8 live accumulators, no stores in between ----
  v8f accs[8];
#pragma unroll
  for (int m = 0; m < 8; ++m) {
    v16bf a0 = *reinterpret_cast<const v16bf*>(&sm.w1s[((m * 2 + 0) * 32 + lane) * 16]);
    v16bf a1 = *reinterpret_cast<const v16bf*>(&sm.w1s[((m * 2 + 1) * 32 + lane) * 16]);
    v8f acc = {};
    acc = __builtin_amdgcn_wmma_f32_16x16x32_bf16(false, a0, false, b0, (short)0, acc, false, false);
    accs[m] = __builtin_amdgcn_wmma_f32_16x16x32_bf16(false, a1, false, b1, (short)0, acc, false, false);
  }
  // ---- Phase B2: drain — ReLU + bf16 pack + hS stores (hazard NOPs amortized) ----
#pragma unroll
  for (int m = 0; m < 8; ++m) {
    // D layout: VGPR jj -> row 16m + jj + 8*hi, col lp
    v8bf hv;
#pragma unroll
    for (int jj = 0; jj < 8; ++jj) {
      float rv = accs[m][jj] > 0.f ? accs[m][jj] : 0.f;
      hv[jj] = (__bf16)rv;
    }
    *reinterpret_cast<v8bf*>(&sm.hS[wave][lp][m * 16 + hi * 8]) = hv;
  }
  __syncthreads();

  // ---- Phase C: layer 2, out = W2 * h (K=128 in 4 steps); W2 frags from global ----
  {
    const __bf16* w2f = wf + 8192;
    v8f acc = {};
#pragma unroll
    for (int kk = 0; kk < 4; ++kk) {
      v16bf a  = *reinterpret_cast<const v16bf*>(w2f + (kk * 32 + lane) * 16);
      v16bf bb = *reinterpret_cast<const v16bf*>(&sm.hS[wave][lp][kk * 32 + hi * 16]);
      acc = __builtin_amdgcn_wmma_f32_16x16x32_bf16(false, a, false, bb, (short)0, acc, false, false);
    }
    const int xw = x0 + lp;
    const int y  = y0 + r;
#pragma unroll
    for (int jj = 0; jj < 8; ++jj) {
      int co = jj + hi * 8;
      out[((b * C + co) * Hn + y) * Wn + xw] = acc[jj];
    }
  }
}

extern "C" void kernel_launch(void* const* d_in, const int* in_sizes, int n_in,
                              void* d_out, int out_size, void* d_ws, size_t ws_size,
                              hipStream_t stream) {
  const float* x   = (const float*)d_in[0];  // [16,16,256,256]
  const float* dna = (const float*)d_in[1];  // [16,4,256,256]
  const float* w1  = (const float*)d_in[2];  // [128,52]
  const float* w2  = (const float*)d_in[3];  // [16,128]
  float* out = (float*)d_out;                // [16,16,256,256]
  __bf16* wf = (__bf16*)d_ws;                // 10240 bf16 = 20480 B of workspace

  // one-shot weight fragment prep (re-run every call: deterministic)
  prep_weights_kernel<<<dim3(40), dim3(256), 0, stream>>>(w1, w2, wf);

  const int blocks = Bn * (Hn / ROWS) * (Wn / WCH);   // 16*32*16 = 8192
  nca_update_kernel<<<dim3(blocks), dim3(256), 0, stream>>>(x, dna, wf, out);
}
